// BertLayer_88467736363857
// MI455X (gfx1250) — compile-verified, matrix operand order checked by reading, not run on hardware
//
#include <hip/hip_runtime.h>
#include <cstdint>
#include <cstddef>

// ---------------------------------------------------------------------------
// BERT layer (B=8, S=512, H=768, NH=12, HD=64, FF=3072) on gfx1250 (CDNA5).
// - All GEMMs + attention matmuls: v_wmma_f32_16x16x32_bf16 (wave32)
// - WMMA fragments load from LDS as 2 x ds_load_b128 (tiles staged K-contig)
// - Global->LDS staging is double-buffered via GLOBAL_LOAD_ASYNC_TO_LDS_B128
//   (ASYNCcnt), overlapping the next K-tile copy with the current WMMAs.
// - Each GEMM wave computes 32x64 (2 A-frags x 4 shared B-frags = 8 WMMA/step)
// ---------------------------------------------------------------------------

#define B_   8
#define S_   512
#define H_   768
#define NH_  12
#define HD_  64
#define FF_  3072
#define M_   (B_ * S_)           // 4096 token rows
#define MAXP 512
#define RELN (2 * MAXP - 1)      // 1023 rel-table rows
#define SCALE_QK 0.125f          // 1/sqrt(64)

typedef __bf16 bf16_t;
typedef __attribute__((ext_vector_type(16))) __bf16 v16bf;
typedef __attribute__((ext_vector_type(8)))  float  v8f;
typedef int v4i __attribute__((vector_size(16)));

// ---- CDNA5 async global->LDS copy (16B per lane), with safe fallback ------
#if defined(__has_builtin)
#if __has_builtin(__builtin_amdgcn_global_load_async_to_lds_b128)
#define ASYNC_LDS 1
#endif
#endif

__device__ __forceinline__ void async_copy16(void* lds, const void* g) {
#ifdef ASYNC_LDS
  typedef __attribute__((address_space(1))) v4i GV;   // global v4i
  typedef __attribute__((address_space(3))) v4i LV;   // LDS v4i
  __builtin_amdgcn_global_load_async_to_lds_b128((GV*)(void*)g, (LV*)lds, 0, 0);
#else
  *(uint4*)lds = *(const uint4*)g;
#endif
}

__device__ __forceinline__ void async_wait() {
#ifdef ASYNC_LDS
#if __has_builtin(__builtin_amdgcn_s_wait_asynccnt)
  __builtin_amdgcn_s_wait_asynccnt(0);
#else
  asm volatile("s_wait_asynccnt 0x0" ::: "memory");
#endif
#endif
}

// ---- WMMA wrapper: D = A(16x32 bf16) x B(32x16 bf16) + C(16x16 f32) -------
__device__ __forceinline__ v8f wmma_bf16(v16bf a, v16bf b, v8f c) {
  return __builtin_amdgcn_wmma_f32_16x16x32_bf16(
      /*neg_a=*/false, a, /*neg_b=*/false, b,
      /*c_mod=*/(short)0, c, /*reuse_a=*/false, /*reuse_b=*/false);
}

// Fragment load (A or B^T pattern). Per ISA 7.12.2, lane (i = lane%16,
// half = lane/16) holds K = {half*8 .. +7} then {16+half*8 .. +7}:
// two contiguous 16B runs when the tile row is K-contiguous in LDS.
__device__ __forceinline__ v16bf load_frag(const bf16_t* s, int ld,
                                           int row0, int k0, int lane) {
  const int r  = row0 + (lane & 15);
  const int kb = k0 + (lane >> 4) * 8;
  const bf16_t* base = s + (size_t)r * ld + kb;
  union { uint4 u[2]; v16bf v; } f;
  f.u[0] = *((const uint4*)base);        // K run 0 (8 bf16)
  f.u[1] = *((const uint4*)(base + 16)); // K run 1 (8 bf16)
  return f.v;
}

// Store a 16x16 f32 accumulator tile to LDS (C layout: m = r + 8*(L/16), n = L%16)
__device__ __forceinline__ void store_C(float* s, int ld, int n0, v8f c, int lane) {
  int n = lane & 15, half = lane >> 4;
#pragma unroll
  for (int r = 0; r < 8; ++r) s[(size_t)(r + 8 * half) * ld + n0 + n] = c[r];
}

// ---------------------------------------------------------------------------
// fp32 -> bf16 conversion (plain)
// ---------------------------------------------------------------------------
__global__ void cvt_bf16_kernel(const float* __restrict__ in,
                                bf16_t* __restrict__ out, int n) {
  int i = blockIdx.x * blockDim.x + threadIdx.x;
  if (i < n) out[i] = (bf16_t)in[i];
}

// fp32 [K][N] -> bf16 transposed [N][K] (for GEMM B operands)
__global__ __launch_bounds__(256)
void cvt_transpose_kernel(const float* __restrict__ in, bf16_t* __restrict__ out,
                          int K, int N) {
  __shared__ float tile[32][33];
  const int kt = blockIdx.y * 32, nt = blockIdx.x * 32;
  const int tx = threadIdx.x & 31, ty = threadIdx.x >> 5; // 32x8
#pragma unroll
  for (int i = 0; i < 32; i += 8)
    tile[ty + i][tx] = in[(size_t)(kt + ty + i) * N + nt + tx];
  __syncthreads();
#pragma unroll
  for (int i = 0; i < 32; i += 8)
    out[(size_t)(nt + ty + i) * K + kt + tx] = (bf16_t)tile[tx][ty + i];
}

// ---------------------------------------------------------------------------
// bf16 GEMM: C[M,N] = A[M,K] * B[K,N] + bias, B supplied TRANSPOSED [N][K].
// Block = 256 threads (8 waves), tile 256x64, K-step 32, double-buffered LDS
// with async global->LDS staging. Each wave: 32x64 = 2 A-frags x 4 B-frags.
// ---------------------------------------------------------------------------
enum { OUT_F32 = 0, OUT_BF16_GELU = 1, OUT_BF16_HEADS = 2, OUT_BF16_HEADS_T = 3 };

__global__ __launch_bounds__(256)
void gemm_bf16_kernel(const bf16_t* __restrict__ A, const bf16_t* __restrict__ BT,
                      const float* __restrict__ bias, void* __restrict__ out,
                      int Mdim, int Ndim, int Kdim, int mode) {
  __shared__ __align__(16) bf16_t As[2][256][32];
  __shared__ __align__(16) bf16_t Bs[2][64][32];   // B^T tile: [n][k]

  const int tid  = threadIdx.x;
  const int lane = tid & 31;
  const int wave = tid >> 5;
  const int bm   = blockIdx.x * 256;
  const int bn   = blockIdx.y * 64;
  const int nk   = Kdim >> 5;

  auto stage = [&](int t, int pb) {
    const int k0 = t * 32;
#pragma unroll
    for (int i = tid; i < 256 * 4; i += 256) {
      int r = i >> 2, c = i & 3;
      async_copy16(&As[pb][r][c * 8], A + (size_t)(bm + r) * Kdim + k0 + c * 8);
    }
    {
      int r = tid >> 2, c = tid & 3;
      async_copy16(&Bs[pb][r][c * 8], BT + (size_t)(bn + r) * Kdim + k0 + c * 8);
    }
  };

  v8f acc0[4] = {};
  v8f acc1[4] = {};

  stage(0, 0);
  async_wait();
  __syncthreads();

  for (int t = 0; t < nk; ++t) {
    const int cur = t & 1;
    if (t + 1 < nk) stage(t + 1, cur ^ 1);   // overlaps with WMMAs below

    // Hoist all fragment loads so ds_loads pipeline, then burst 8 WMMAs.
    v16bf af0 = load_frag(&As[cur][0][0], 32, wave * 32,      0, lane);
    v16bf af1 = load_frag(&As[cur][0][0], 32, wave * 32 + 16, 0, lane);
    v16bf bf[4];
#pragma unroll
    for (int j = 0; j < 4; ++j) bf[j] = load_frag(&Bs[cur][0][0], 32, j * 16, 0, lane);
#pragma unroll
    for (int j = 0; j < 4; ++j) acc0[j] = wmma_bf16(af0, bf[j], acc0[j]);
#pragma unroll
    for (int j = 0; j < 4; ++j) acc1[j] = wmma_bf16(af1, bf[j], acc1[j]);

    async_wait();       // next tile fully in LDS
    __syncthreads();    // all waves done reading 'cur', writes visible
  }

  // Epilogue
  const int nl = lane & 15, hf = lane >> 4;
#pragma unroll
  for (int g = 0; g < 2; ++g) {
    v8f* acc = g ? acc1 : acc0;
#pragma unroll
    for (int j = 0; j < 4; ++j) {
#pragma unroll
      for (int r = 0; r < 8; ++r) {
        int m = bm + wave * 32 + g * 16 + r + 8 * hf;
        int n = bn + j * 16 + nl;
        float v = acc[j][r] + bias[n];
        if (mode == OUT_F32) {
          ((float*)out)[(size_t)m * Ndim + n] = v;
        } else if (mode == OUT_BF16_GELU) {
          float gl = 0.5f * v * (1.0f + erff(v * 0.7071067811865475f));
          ((bf16_t*)out)[(size_t)m * Ndim + n] = (bf16_t)gl;
        } else if (mode == OUT_BF16_HEADS) {      // [B, NH, S, HD]
          int b = m / S_, s = m % S_;
          int h = n / HD_, d = n % HD_;
          ((bf16_t*)out)[((((size_t)b * NH_ + h) * S_) + s) * HD_ + d] = (bf16_t)v;
        } else {                                  // OUT_BF16_HEADS_T: [B, NH, HD, S]
          int b = m / S_, s = m % S_;
          int h = n / HD_, d = n % HD_;
          ((bf16_t*)out)[((((size_t)b * NH_ + h) * HD_) + d) * S_ + s] = (bf16_t)v;
        }
      }
    }
  }
}

// ---------------------------------------------------------------------------
// Flash attention with relative-position bias, one wave per (b, h, 16 q-rows).
// bias[m,n] = T[m, m-n+31] with T = Qtile x RelSlice^T  (WMMA GEMM + gather)
// K/V/rel tiles double-buffered with async staging.
// ---------------------------------------------------------------------------
__global__ __launch_bounds__(32)
void attention_kernel(const bf16_t* __restrict__ Qb, const bf16_t* __restrict__ Kb,
                      const bf16_t* __restrict__ Vb, const bf16_t* __restrict__ relb,
                      const unsigned char* __restrict__ mask,
                      bf16_t* __restrict__ ctx) {
  __shared__ __align__(16) bf16_t Qs[16][HD_];
  __shared__ __align__(16) bf16_t Kt[2][32][HD_];   // [pos][d]
  __shared__ __align__(16) bf16_t Vt[2][HD_][32];   // [d][pos]
  __shared__ __align__(16) bf16_t Rs[2][48][HD_];   // [diag][d]
  __shared__ __align__(16) bf16_t Pb[16][32];       // probs [m][n]
  __shared__ float  Tb[16][48];
  __shared__ float  Sb[16][32];
  __shared__ float  mrow[16], lrow[16], arow[16];

  const int lane = threadIdx.x;
  const int qt = blockIdx.x, h = blockIdx.y, b = blockIdx.z;
  const int q0 = qt * 16;

  const bf16_t* Qh = Qb + (((size_t)b * NH_ + h) * S_) * HD_;
  const bf16_t* Kh = Kb + (((size_t)b * NH_ + h) * S_) * HD_;
  const bf16_t* Vh = Vb + (((size_t)b * NH_ + h) * HD_) * S_;  // head-transposed

  auto stage = [&](int t, int pb) {
    const int k0 = t * 32;
    for (int i = lane; i < 32 * 8; i += 32) {          // K tile [32][64]
      int r = i >> 3, c = i & 7;
      async_copy16(&Kt[pb][r][c * 8], Kh + (size_t)(k0 + r) * HD_ + c * 8);
    }
    for (int i = lane; i < 64 * 4; i += 32) {          // V tile [64][32]
      int r = i >> 2, c = i & 3;
      async_copy16(&Vt[pb][r][c * 8], Vh + (size_t)r * S_ + k0 + c * 8);
    }
    const int base = q0 - k0 + (MAXP - 1) - 31;        // 48 rel diagonals
    for (int j = lane; j < 48; j += 32) {
      int idx = base + j;
      idx = idx < 0 ? 0 : (idx > RELN - 1 ? RELN - 1 : idx);
#pragma unroll
      for (int c = 0; c < 8; ++c)
        async_copy16(&Rs[pb][j][c * 8], relb + (size_t)idx * HD_ + c * 8);
    }
  };

  // Stage Q tile (16x64)
  for (int i = lane; i < 16 * 8; i += 32) {
    int r = i >> 3, c = i & 7;
    *((uint4*)&Qs[r][c * 8]) = *((const uint4*)(Qh + (size_t)(q0 + r) * HD_ + c * 8));
  }
  if (lane < 16) { mrow[lane] = -3.0e38f; lrow[lane] = 0.0f; arow[lane] = 0.0f; }

  stage(0, 0);
  async_wait();
  __syncthreads();

  const v16bf aq0 = load_frag(&Qs[0][0], HD_, 0, 0,  lane);
  const v16bf aq1 = load_frag(&Qs[0][0], HD_, 0, 32, lane);

  v8f o[4] = {};
  const int NCHUNK = S_ / 32;

  for (int t = 0; t < NCHUNK; ++t) {
    const int cur = t & 1;
    const int k0 = t * 32;
    if (t + 1 < NCHUNK) stage(t + 1, cur ^ 1);  // async, overlaps compute

    // S-tile (16x32) = Q x K^T ; T-tile (16x48) = Q x RelSlice^T
    v8f sc[2] = {};
    v8f tf[3] = {};
#pragma unroll
    for (int hd = 0; hd < 2; ++hd) {
      v16bf aq = hd ? aq1 : aq0;
      v16bf bk0 = load_frag(&Kt[cur][0][0], HD_, 0,  hd * 32, lane);
      v16bf bk1 = load_frag(&Kt[cur][0][0], HD_, 16, hd * 32, lane);
      v16bf br0 = load_frag(&Rs[cur][0][0], HD_, 0,  hd * 32, lane);
      v16bf br1 = load_frag(&Rs[cur][0][0], HD_, 16, hd * 32, lane);
      v16bf br2 = load_frag(&Rs[cur][0][0], HD_, 32, hd * 32, lane);
      sc[0] = wmma_bf16(aq, bk0, sc[0]);
      sc[1] = wmma_bf16(aq, bk1, sc[1]);
      tf[0] = wmma_bf16(aq, br0, tf[0]);
      tf[1] = wmma_bf16(aq, br1, tf[1]);
      tf[2] = wmma_bf16(aq, br2, tf[2]);
    }
#pragma unroll
    for (int j = 0; j < 2; ++j) store_C(&Sb[0][0], 32, j * 16, sc[j], lane);
#pragma unroll
    for (int j = 0; j < 3; ++j) store_C(&Tb[0][0], 48, j * 16, tf[j], lane);
    __syncthreads();

    // Online softmax, lanes 0..15 each own one q-row.
    if (lane < 16) {
      int m = lane;
      float mx = mrow[m];
      float sreg[32];
#pragma unroll
      for (int n = 0; n < 32; ++n) {
        float v = Sb[m][n] * SCALE_QK + Tb[m][m - n + 31];
        if (!mask[(size_t)b * S_ + k0 + n]) v = -3.0e38f;
        sreg[n] = v;
        mx = fmaxf(mx, v);
      }
      float alpha = __expf(mrow[m] - mx);
      float sum = 0.0f;
#pragma unroll
      for (int n = 0; n < 32; ++n) {
        float p = __expf(sreg[n] - mx);
        sum += p;
        Pb[m][n] = (bf16_t)p;
      }
      mrow[m] = mx;
      lrow[m] = lrow[m] * alpha + sum;
      arow[m] = alpha;
    }
    __syncthreads();

    // O = O * alpha + P x V   (P: A frag 16x32; V^T tile gives 4 B frags)
    {
      const int hf = lane >> 4;
      v16bf ap = load_frag(&Pb[0][0], 32, 0, 0, lane);
      v16bf bv[4];
#pragma unroll
      for (int j = 0; j < 4; ++j) bv[j] = load_frag(&Vt[cur][0][0], 32, j * 16, 0, lane);
#pragma unroll
      for (int j = 0; j < 4; ++j) {
#pragma unroll
        for (int r = 0; r < 8; ++r) o[j][r] *= arow[r + 8 * hf];
        o[j] = wmma_bf16(ap, bv[j], o[j]);
      }
    }
    async_wait();      // next chunk staged
    __syncthreads();
  }

  // Epilogue: normalize and write context in [B, S, H] bf16 (heads re-gathered)
  const int nl = lane & 15, hf = lane >> 4;
#pragma unroll
  for (int j = 0; j < 4; ++j) {
#pragma unroll
    for (int r = 0; r < 8; ++r) {
      int m = r + 8 * hf;
      int q = q0 + m;
      int d = j * 16 + nl;
      float v = o[j][r] / lrow[m];
      ctx[((size_t)b * S_ + q) * H_ + h * HD_ + d] = (bf16_t)v;
    }
  }
}

// ---------------------------------------------------------------------------
// Residual add + LayerNorm over H=768; optional bf16 mirror for next GEMM.
// ---------------------------------------------------------------------------
__global__ __launch_bounds__(256)
void ln_kernel(const float* __restrict__ x, const float* __restrict__ resid,
               const float* __restrict__ scale, const float* __restrict__ bias,
               float* __restrict__ out, bf16_t* __restrict__ out_bf) {
  __shared__ float red[256];
  __shared__ float red2[256];
  const int row = blockIdx.x;
  const float* xr = x + (size_t)row * H_;
  const float* rr = resid + (size_t)row * H_;

  float local[3];
  float s = 0.0f, s2 = 0.0f;
#pragma unroll
  for (int i = 0; i < 3; ++i) {
    int c = threadIdx.x + i * 256;
    float v = xr[c] + rr[c];
    local[i] = v;
    s += v;
    s2 += v * v;
  }
  red[threadIdx.x] = s;
  red2[threadIdx.x] = s2;
  __syncthreads();
  for (int off = 128; off > 0; off >>= 1) {
    if (threadIdx.x < off) {
      red[threadIdx.x]  += red[threadIdx.x + off];
      red2[threadIdx.x] += red2[threadIdx.x + off];
    }
    __syncthreads();
  }
  const float mu  = red[0] * (1.0f / H_);
  const float var = red2[0] * (1.0f / H_) - mu * mu;
  const float inv = rsqrtf(var + 1e-12f);
#pragma unroll
  for (int i = 0; i < 3; ++i) {
    int c = threadIdx.x + i * 256;
    float v = (local[i] - mu) * inv * scale[c] + bias[c];
    out[(size_t)row * H_ + c] = v;
    if (out_bf) out_bf[(size_t)row * H_ + c] = (bf16_t)v;
  }
}

// ---------------------------------------------------------------------------
// Host-side orchestration
// ---------------------------------------------------------------------------
extern "C" void kernel_launch(void* const* d_in, const int* in_sizes, int n_in,
                              void* d_out, int out_size, void* d_ws, size_t ws_size,
                              hipStream_t stream) {
  (void)in_sizes; (void)n_in; (void)out_size; (void)ws_size;

  const float* hidden = (const float*)d_in[0];
  const unsigned char* mask = (const unsigned char*)d_in[1];
  const float* Wq = (const float*)d_in[2];  const float* bq = (const float*)d_in[3];
  const float* Wk = (const float*)d_in[4];  const float* bk = (const float*)d_in[5];
  const float* Wv = (const float*)d_in[6];  const float* bv = (const float*)d_in[7];
  const float* Wo = (const float*)d_in[8];  const float* bo = (const float*)d_in[9];
  const float* rel = (const float*)d_in[10];
  const float* ln1s = (const float*)d_in[11]; const float* ln1b = (const float*)d_in[12];
  const float* W1 = (const float*)d_in[13]; const float* b1 = (const float*)d_in[14];
  const float* W2 = (const float*)d_in[15]; const float* b2 = (const float*)d_in[16];
  const float* ln2s = (const float*)d_in[17]; const float* ln2b = (const float*)d_in[18];
  float* out = (float*)d_out;

  // --- workspace carve-up (256B aligned) ---
  char* ws = (char*)d_ws;
  size_t off = 0;
  auto alloc = [&](size_t bytes) -> char* {
    char* p = ws + off;
    off += (bytes + 255) & ~(size_t)255;
    return p;
  };
  bf16_t* Xb   = (bf16_t*)alloc((size_t)M_ * H_ * 2);     // token activations bf16
  bf16_t* Qb   = (bf16_t*)alloc((size_t)M_ * H_ * 2);     // [B,NH,S,HD]
  bf16_t* Kb   = (bf16_t*)alloc((size_t)M_ * H_ * 2);     // [B,NH,S,HD]
  bf16_t* Vb   = (bf16_t*)alloc((size_t)M_ * H_ * 2);     // [B,NH,HD,S]
  bf16_t* ffb  = (bf16_t*)Xb;                             // aliases Xb..Vb (exact fit)
  bf16_t* ctxb = (bf16_t*)alloc((size_t)M_ * H_ * 2);
  float*  tmp  = (float*) alloc((size_t)M_ * H_ * 4);
  float*  tmp2 = tmp;                                     // aliases tmp (dead by then)
  float*  hid  = (float*) alloc((size_t)M_ * H_ * 4);
  bf16_t* hidb = (bf16_t*)alloc((size_t)M_ * H_ * 2);
  bf16_t* WqT  = (bf16_t*)alloc((size_t)H_ * H_ * 2);     // transposed bf16 weights
  bf16_t* WkT  = (bf16_t*)alloc((size_t)H_ * H_ * 2);
  bf16_t* WvT  = (bf16_t*)alloc((size_t)H_ * H_ * 2);
  bf16_t* WoT  = (bf16_t*)alloc((size_t)H_ * H_ * 2);
  bf16_t* W1T  = (bf16_t*)alloc((size_t)H_ * FF_ * 2);    // [FF][H]
  bf16_t* W2T  = (bf16_t*)alloc((size_t)FF_ * H_ * 2);    // [H][FF]
  bf16_t* relb = (bf16_t*)alloc((size_t)RELN * HD_ * 2);

  cvt_bf16_kernel<<<(M_ * H_ + 255) / 256, 256, 0, stream>>>(hidden, Xb, M_ * H_);
  cvt_bf16_kernel<<<(RELN * HD_ + 255) / 256, 256, 0, stream>>>(rel, relb, RELN * HD_);

  auto cvtT = [&](const float* src, bf16_t* dst, int K, int N) {
    dim3 g(N / 32, K / 32);
    cvt_transpose_kernel<<<g, 256, 0, stream>>>(src, dst, K, N);
  };
  cvtT(Wq, WqT, H_, H_);
  cvtT(Wk, WkT, H_, H_);
  cvtT(Wv, WvT, H_, H_);
  cvtT(Wo, WoT, H_, H_);
  cvtT(W1, W1T, H_, FF_);
  cvtT(W2, W2T, FF_, H_);

  // QKV projections
  dim3 gq(M_ / 256, H_ / 64);
  gemm_bf16_kernel<<<gq, 256, 0, stream>>>(Xb, WqT, bq, Qb, M_, H_, H_, OUT_BF16_HEADS);
  gemm_bf16_kernel<<<gq, 256, 0, stream>>>(Xb, WkT, bk, Kb, M_, H_, H_, OUT_BF16_HEADS);
  gemm_bf16_kernel<<<gq, 256, 0, stream>>>(Xb, WvT, bv, Vb, M_, H_, H_, OUT_BF16_HEADS_T);

  // Flash attention with relative-position bias -> ctx [B,S,H] bf16
  dim3 ga(S_ / 16, NH_, B_);
  attention_kernel<<<ga, 32, 0, stream>>>(Qb, Kb, Vb, relb, mask, ctxb);

  // Output projection -> tmp f32
  gemm_bf16_kernel<<<gq, 256, 0, stream>>>(ctxb, WoT, bo, tmp, M_, H_, H_, OUT_F32);

  // LN1(residual + attn_out) -> hid f32 + hidb bf16
  ln_kernel<<<M_, 256, 0, stream>>>(tmp, hidden, ln1s, ln1b, hid, hidb);

  // FFN: GELU(hid @ W1 + b1) bf16 -> @ W2 + b2 f32
  dim3 gf1(M_ / 256, FF_ / 64);
  gemm_bf16_kernel<<<gf1, 256, 0, stream>>>(hidb, W1T, b1, ffb, M_, FF_, H_, OUT_BF16_GELU);
  dim3 gf2(M_ / 256, H_ / 64);
  gemm_bf16_kernel<<<gf2, 256, 0, stream>>>(ffb, W2T, b2, tmp2, M_, H_, FF_, OUT_F32);

  // LN2(hid + ff) -> final output f32
  ln_kernel<<<M_, 256, 0, stream>>>(tmp2, hid, ln2s, ln2b, out, nullptr);
}
